// BaselineGAT_43276090474726
// MI455X (gfx1250) — compile-verified, matrix operand order checked by reading, not run on hardware
//
#include <hip/hip_runtime.h>
#include <hip/hip_bf16.h>

typedef __attribute__((ext_vector_type(16))) __bf16 v16bf;
typedef __attribute__((ext_vector_type(8)))  __bf16 v8bf;
typedef __attribute__((ext_vector_type(8)))  float  v8f;

#define IN_DIM 1488
#define HID    1024
#define OUTD   512
#define CH     32
#define NH     8
#define FDIM   256   // NH*CH

// ---------------------------------------------------------------------------
// LayerNorm (f32 in) -> bf16 out
// ---------------------------------------------------------------------------
__global__ __launch_bounds__(256) void layernorm_cast_kernel(
    const float* __restrict__ x, const float* __restrict__ lng,
    const float* __restrict__ lnb, __bf16* __restrict__ xn, int K)
{
    const int row = blockIdx.x;
    const float* xr = x + (size_t)row * K;
    float s = 0.f, s2 = 0.f;
    for (int i = threadIdx.x; i < K; i += 256) {
        float v = xr[i]; s += v; s2 += v * v;
    }
    __shared__ float rs[256], rq[256];
    rs[threadIdx.x] = s; rq[threadIdx.x] = s2;
    __syncthreads();
    for (int off = 128; off > 0; off >>= 1) {
        if (threadIdx.x < off) {
            rs[threadIdx.x] += rs[threadIdx.x + off];
            rq[threadIdx.x] += rq[threadIdx.x + off];
        }
        __syncthreads();
    }
    const float mu   = rs[0] / (float)K;
    const float var  = rq[0] / (float)K - mu * mu;
    const float rstd = rsqrtf(var + 1e-5f);
    __bf16* xo = xn + (size_t)row * K;
    for (int i = threadIdx.x; i < K; i += 256) {
        float v = (xr[i] - mu) * rstd * lng[i] + lnb[i];
        xo[i] = (__bf16)v;
    }
}

// ---------------------------------------------------------------------------
// Cast + transpose f32 [K,N] -> bf16 [N,K]
// ---------------------------------------------------------------------------
__global__ __launch_bounds__(256) void cast_transpose_kernel(
    const float* __restrict__ W, __bf16* __restrict__ Wt, int K, int Ncol)
{
    int idx = blockIdx.x * 256 + threadIdx.x;
    if (idx < K * Ncol) {
        int k = idx / Ncol, n = idx % Ncol;
        Wt[(size_t)n * K + k] = (__bf16)W[idx];
    }
}

// ---------------------------------------------------------------------------
// bf16 WMMA GEMM: C[M,Ncol] = A[M,K] * Bt[Ncol,K]^T (+bias) (+relu)
// 256 threads = 8 waves (4x2); block tile 128x128; wave tile 32x64 (8 accums)
// Double-buffered LDS fed by global_load_async_to_lds_b128 (ASYNCcnt).
// ---------------------------------------------------------------------------
template <bool RELU, bool STORE_BF16, bool HAS_BIAS>
__global__ __launch_bounds__(256) void gemm_bf16_wmma_kernel(
    const __bf16* __restrict__ A, const __bf16* __restrict__ Bt,
    const float* __restrict__ bias, float* __restrict__ Cf,
    __bf16* __restrict__ Cb, int M, int Ncol, int K)
{
    constexpr int BM = 128, BN = 128, BK = 32, PITCH = BK + 8;
    __shared__ __bf16 sA[2][BM][PITCH];
    __shared__ __bf16 sB[2][BN][PITCH];

    const int tid  = threadIdx.x;
    const int wave = tid >> 5;
    const int lane = tid & 31;
    const int wm   = wave >> 1;          // 0..3 (M direction, 32 rows each)
    const int wn   = wave & 1;           // 0..1 (N direction, 64 cols each)
    const int m0   = blockIdx.y * BM;
    const int n0   = blockIdx.x * BN;

    const int laneRow  = lane & 15;
    const int laneHalf = lane >> 4;      // 0 or 1
    const int kb       = laneHalf * 8;   // K sub-chunk base (0 or 8)

    const bool interior = (m0 + BM <= M) && (n0 + BN <= Ncol);

    // fast stage: 4 x global_load_async_to_lds_b128 per thread (no guards)
    auto stage_fast = [&](int buf, int k0) {
        #pragma unroll
        for (int i = 0; i < 2; ++i) {
            const int cid = tid + i * 256;          // 0..511
            const int r   = cid >> 2;               // 0..127
            const int cg  = (cid & 3) * 8;          // 0,8,16,24
            {
                unsigned           lo = (unsigned)(size_t)&sA[buf][r][cg];
                unsigned long long ga = (unsigned long long)(size_t)
                    (A + (size_t)(m0 + r) * K + (k0 + cg));
                asm volatile("global_load_async_to_lds_b128 %0, %1, off"
                             :: "v"(lo), "v"(ga) : "memory");
            }
            {
                unsigned           lo = (unsigned)(size_t)&sB[buf][r][cg];
                unsigned long long ga = (unsigned long long)(size_t)
                    (Bt + (size_t)(n0 + r) * K + (k0 + cg));
                asm volatile("global_load_async_to_lds_b128 %0, %1, off"
                             :: "v"(lo), "v"(ga) : "memory");
            }
        }
    };

    // slow stage: guarded loads, zero-pad OOB, ds_store
    auto stage_slow = [&](int buf, int k0) {
        #pragma unroll
        for (int i = 0; i < 2; ++i) {
            const int cid = tid + i * 256;
            const int r   = cid >> 2;
            const int cg  = (cid & 3) * 8;
            const int gk  = k0 + cg;
            v8bf va = v8bf{};
            if ((m0 + r) < M && gk + 8 <= K)
                va = *(const v8bf*)(A + (size_t)(m0 + r) * K + gk);
            *(v8bf*)&sA[buf][r][cg] = va;
            v8bf vb = v8bf{};
            if ((n0 + r) < Ncol && gk + 8 <= K)
                vb = *(const v8bf*)(Bt + (size_t)(n0 + r) * K + gk);
            *(v8bf*)&sB[buf][r][cg] = vb;
        }
    };

    v8f acc[2][4] = {};

    int  buf     = 0;
    bool curFast = interior && (BK <= K);
    if (curFast) stage_fast(0, 0); else stage_slow(0, 0);

    for (int k0 = 0; k0 < K; k0 += BK) {
        const int  kn      = k0 + BK;
        const bool hasNext = kn < K;
        bool nextFast = false;
        if (hasNext) {
            nextFast = interior && (kn + BK <= K);
            if (nextFast) stage_fast(buf ^ 1, kn);
            else          stage_slow(buf ^ 1, kn);
        }
        if (curFast || nextFast) {
            if (hasNext && nextFast)
                asm volatile("s_wait_asynccnt 0x4" ::: "memory");
            else
                asm volatile("s_wait_asynccnt 0x0" ::: "memory");
        }
        __syncthreads();

        // A fragments (16-bit A 16x32 layout: lanes 0-15 K 0..7/16..23)
        v16bf afrag[2];
        #pragma unroll
        for (int rt = 0; rt < 2; ++rt) {
            const int ar = wm * 32 + rt * 16 + laneRow;
            v8bf alo = *(const v8bf*)&sA[buf][ar][kb];
            v8bf ahi = *(const v8bf*)&sA[buf][ar][kb + 16];
            afrag[rt] = __builtin_shufflevector(alo, ahi,
                0, 1, 2, 3, 4, 5, 6, 7, 8, 9, 10, 11, 12, 13, 14, 15);
        }
        #pragma unroll
        for (int ct = 0; ct < 4; ++ct) {
            const int bc = wn * 64 + ct * 16 + laneRow;
            v8bf blo = *(const v8bf*)&sB[buf][bc][kb];
            v8bf bhi = *(const v8bf*)&sB[buf][bc][kb + 16];
            v16bf bfrag = __builtin_shufflevector(blo, bhi,
                0, 1, 2, 3, 4, 5, 6, 7, 8, 9, 10, 11, 12, 13, 14, 15);
            #pragma unroll
            for (int rt = 0; rt < 2; ++rt)
                acc[rt][ct] = __builtin_amdgcn_wmma_f32_16x16x32_bf16(
                    false, afrag[rt], false, bfrag, (short)0, acc[rt][ct],
                    false, false);
        }
        __syncthreads();
        buf ^= 1;
        curFast = nextFast;
    }

    // store: C 16x16 f32 layout: N = lane&15, M = r + 8*(lane>>4)
    #pragma unroll
    for (int rt = 0; rt < 2; ++rt) {
        const int rowBase = m0 + wm * 32 + rt * 16 + laneHalf * 8;
        #pragma unroll
        for (int ct = 0; ct < 4; ++ct) {
            const int col = n0 + wn * 64 + ct * 16 + laneRow;
            if (col >= Ncol) continue;
            const float bv = HAS_BIAS ? bias[col] : 0.0f;
            #pragma unroll
            for (int r = 0; r < 8; ++r) {
                const int row = rowBase + r;
                if (row >= M) continue;
                float v = acc[rt][ct][r] + bv;
                if (RELU) v = v > 0.f ? v : 0.f;
                if (STORE_BF16) Cb[(size_t)row * Ncol + col] = (__bf16)v;
                else            Cf[(size_t)row * Ncol + col] = v;
            }
        }
    }
}

// ---------------------------------------------------------------------------
// Attention scores: a_src/a_dst [N,H] from g [N,H,C]
// ---------------------------------------------------------------------------
__global__ __launch_bounds__(256) void att_scores_kernel(
    const float* __restrict__ g, const float* __restrict__ att_src,
    const float* __restrict__ att_dst, float* __restrict__ asrc,
    float* __restrict__ adst, int N)
{
    int idx = blockIdx.x * 256 + threadIdx.x;
    if (idx >= N * NH) return;
    const int n = idx / NH, h = idx % NH;
    const float* gp = g + (size_t)n * FDIM + h * CH;
    float ss = 0.f, sd = 0.f;
    #pragma unroll
    for (int c = 0; c < CH; ++c) {
        float v = gp[c];
        ss += v * att_src[h * CH + c];
        sd += v * att_dst[h * CH + c];
    }
    asrc[idx] = ss;
    adst[idx] = sd;
}

__global__ __launch_bounds__(256) void fill_f32_kernel(float* p, float v, size_t n)
{
    size_t i = (size_t)blockIdx.x * 256 + threadIdx.x;
    if (i < n) p[i] = v;
}

__device__ __forceinline__ void atomicMaxF(float* addr, float val)
{
    if (val >= 0.f) atomicMax((int*)addr, __float_as_int(val));
    else            atomicMin((unsigned int*)addr, __float_as_uint(val));
}

__device__ __forceinline__ void edge_endpoints(
    const long long* __restrict__ ei, long long Ecnt, long long e,
    long long& s, long long& d)
{
    if (e < Ecnt) { s = ei[e]; d = ei[Ecnt + e]; }
    else          { s = e - Ecnt; d = e - Ecnt; }
}

// pass 1: per-(dst,head) max of leaky_relu(a_src[src]+a_dst[dst])
__global__ __launch_bounds__(256) void edge_max_kernel(
    const long long* __restrict__ ei, long long Ecnt, int N,
    const float* __restrict__ asrc, const float* __restrict__ adst,
    float* __restrict__ m)
{
    long long idx = (long long)blockIdx.x * 256 + threadIdx.x;
    long long total = (Ecnt + N) * NH;
    if (idx >= total) return;
    long long e = idx / NH; int h = (int)(idx % NH);
    long long s, d; edge_endpoints(ei, Ecnt, e, s, d);
    float v = asrc[s * NH + h] + adst[d * NH + h];
    v = v > 0.f ? v : 0.2f * v;
    atomicMaxF(&m[d * NH + h], v);
}

// pass 2: denom = sum exp(e - m)
__global__ __launch_bounds__(256) void edge_sum_kernel(
    const long long* __restrict__ ei, long long Ecnt, int N,
    const float* __restrict__ asrc, const float* __restrict__ adst,
    const float* __restrict__ m, float* __restrict__ denom)
{
    long long idx = (long long)blockIdx.x * 256 + threadIdx.x;
    long long total = (Ecnt + N) * NH;
    if (idx >= total) return;
    long long e = idx / NH; int h = (int)(idx % NH);
    long long s, d; edge_endpoints(ei, Ecnt, e, s, d);
    float v = asrc[s * NH + h] + adst[d * NH + h];
    v = v > 0.f ? v : 0.2f * v;
    atomicAdd(&denom[d * NH + h], __expf(v - m[d * NH + h]));
}

// pass 3: agg[dst] += alpha * g[src]
__global__ __launch_bounds__(256) void edge_agg_kernel(
    const long long* __restrict__ ei, long long Ecnt, int N,
    const float* __restrict__ asrc, const float* __restrict__ adst,
    const float* __restrict__ m, const float* __restrict__ denom,
    const float* __restrict__ g, float* __restrict__ agg)
{
    long long idx = (long long)blockIdx.x * 256 + threadIdx.x;
    long long total = (Ecnt + N) * NH;
    if (idx >= total) return;
    long long e = idx / NH; int h = (int)(idx % NH);
    long long s, d; edge_endpoints(ei, Ecnt, e, s, d);
    float v = asrc[s * NH + h] + adst[d * NH + h];
    v = v > 0.f ? v : 0.2f * v;
    const float alpha = __expf(v - m[d * NH + h]) / (denom[d * NH + h] + 1e-16f);
    const float* gs = g + (size_t)s * FDIM + h * CH;
    float* ag = agg + (size_t)d * FDIM + h * CH;
    #pragma unroll
    for (int c = 0; c < CH; ++c)
        atomicAdd(&ag[c], alpha * gs[c]);
}

// out = elu(agg + bg) + res
__global__ __launch_bounds__(256) void finalize_kernel(
    const float* __restrict__ agg, const float* __restrict__ bg,
    const float* __restrict__ res, float* __restrict__ out, size_t total)
{
    size_t i = (size_t)blockIdx.x * 256 + threadIdx.x;
    if (i >= total) return;
    const int j = (int)(i % FDIM);
    float v = agg[i] + bg[j];
    v = v > 0.f ? v : (__expf(v) - 1.f);
    out[i] = v + res[i];
}

// ---------------------------------------------------------------------------
// launch
// ---------------------------------------------------------------------------
extern "C" void kernel_launch(void* const* d_in, const int* in_sizes, int n_in,
                              void* d_out, int out_size, void* d_ws, size_t ws_size,
                              hipStream_t stream)
{
    const float*     x    = (const float*)d_in[0];
    const long long* ei   = (const long long*)d_in[1];
    const float*     lng  = (const float*)d_in[2];
    const float*     lnb  = (const float*)d_in[3];
    const float*     W1   = (const float*)d_in[4];
    const float*     b1   = (const float*)d_in[5];
    const float*     W2   = (const float*)d_in[6];
    const float*     b2   = (const float*)d_in[7];
    const float*     Wr   = (const float*)d_in[8];
    const float*     br   = (const float*)d_in[9];
    const float*     Wg   = (const float*)d_in[10];
    const float*     asv  = (const float*)d_in[11];
    const float*     adv  = (const float*)d_in[12];
    const float*     bg   = (const float*)d_in[13];
    float* out = (float*)d_out;

    const int       N = in_sizes[0] / IN_DIM;
    const long long E = in_sizes[1] / 2;

    // workspace carve-out (256B aligned)
    char*  base = (char*)d_ws;
    size_t off  = 0;
    auto alloc = [&](size_t bytes) -> char* {
        char* p = base + off;
        off = (off + bytes + 255) & ~(size_t)255;
        return p;
    };
    __bf16* xn    = (__bf16*)alloc((size_t)N * IN_DIM * 2);
    __bf16* w1t   = (__bf16*)alloc((size_t)IN_DIM * HID * 2);
    __bf16* w2t   = (__bf16*)alloc((size_t)HID * OUTD * 2);
    __bf16* wrt   = (__bf16*)alloc((size_t)OUTD * FDIM * 2);
    __bf16* wgt   = (__bf16*)alloc((size_t)OUTD * FDIM * 2);
    __bf16* h1b   = (__bf16*)alloc((size_t)N * HID * 2);
    __bf16* hb    = (__bf16*)alloc((size_t)N * OUTD * 2);
    float*  res   = (float*)alloc((size_t)N * FDIM * 4);
    float*  gbuf  = (float*)alloc((size_t)N * FDIM * 4);
    float*  asrc  = (float*)alloc((size_t)N * NH * 4);
    float*  adst  = (float*)alloc((size_t)N * NH * 4);
    float*  mbuf  = (float*)alloc((size_t)N * NH * 4);
    float*  denom = (float*)alloc((size_t)N * NH * 4);
    float*  agg   = (float*)alloc((size_t)N * FDIM * 4);

    // 1) LayerNorm -> bf16
    layernorm_cast_kernel<<<N, 256, 0, stream>>>(x, lng, lnb, xn, IN_DIM);

    // 2) weight cast + transpose
    {
        int n1 = IN_DIM * HID, n2 = HID * OUTD, n3 = OUTD * FDIM;
        cast_transpose_kernel<<<(n1 + 255) / 256, 256, 0, stream>>>(W1, w1t, IN_DIM, HID);
        cast_transpose_kernel<<<(n2 + 255) / 256, 256, 0, stream>>>(W2, w2t, HID, OUTD);
        cast_transpose_kernel<<<(n3 + 255) / 256, 256, 0, stream>>>(Wr, wrt, OUTD, FDIM);
        cast_transpose_kernel<<<(n3 + 255) / 256, 256, 0, stream>>>(Wg, wgt, OUTD, FDIM);
    }

    // 3) GEMM chain (WMMA bf16, async double-buffered)
    dim3 blk(256);
    {   // h1 = relu(xn @ W1 + b1), bf16
        dim3 grid((HID + 127) / 128, (N + 127) / 128);
        gemm_bf16_wmma_kernel<true, true, true><<<grid, blk, 0, stream>>>(
            xn, w1t, b1, nullptr, h1b, N, HID, IN_DIM);
    }
    {   // h = h1 @ W2 + b2, bf16
        dim3 grid((OUTD + 127) / 128, (N + 127) / 128);
        gemm_bf16_wmma_kernel<false, true, true><<<grid, blk, 0, stream>>>(
            h1b, w2t, b2, nullptr, hb, N, OUTD, HID);
    }
    {   // res = h @ Wr + br, f32
        dim3 grid((FDIM + 127) / 128, (N + 127) / 128);
        gemm_bf16_wmma_kernel<false, false, true><<<grid, blk, 0, stream>>>(
            hb, wrt, br, res, nullptr, N, FDIM, OUTD);
    }
    {   // g = h @ Wg, f32
        dim3 grid((FDIM + 127) / 128, (N + 127) / 128);
        gemm_bf16_wmma_kernel<false, false, false><<<grid, blk, 0, stream>>>(
            hb, wgt, nullptr, gbuf, nullptr, N, FDIM, OUTD);
    }

    // 4) attention scores
    att_scores_kernel<<<(N * NH + 255) / 256, 256, 0, stream>>>(
        gbuf, asv, adv, asrc, adst, N);

    // 5) init softmax state
    {
        size_t nh = (size_t)N * NH, nf = (size_t)N * FDIM;
        fill_f32_kernel<<<(unsigned)((nh + 255) / 256), 256, 0, stream>>>(mbuf, -INFINITY, nh);
        fill_f32_kernel<<<(unsigned)((nh + 255) / 256), 256, 0, stream>>>(denom, 0.f, nh);
        fill_f32_kernel<<<(unsigned)((nf + 255) / 256), 256, 0, stream>>>(agg, 0.f, nf);
    }

    // 6) edge softmax + aggregation
    {
        long long total = (E + N) * NH;
        unsigned nblk = (unsigned)((total + 255) / 256);
        edge_max_kernel<<<nblk, 256, 0, stream>>>(ei, E, N, asrc, adst, mbuf);
        edge_sum_kernel<<<nblk, 256, 0, stream>>>(ei, E, N, asrc, adst, mbuf, denom);
        edge_agg_kernel<<<nblk, 256, 0, stream>>>(ei, E, N, asrc, adst, mbuf, denom, gbuf, agg);
    }

    // 7) out = elu(agg + bg) + res
    {
        size_t total = (size_t)N * FDIM;
        finalize_kernel<<<(unsigned)((total + 255) / 256), 256, 0, stream>>>(
            agg, bg, res, out, total);
    }
}
